// FP8Linear_23965917512381
// MI455X (gfx1250) — matrix operand order repaired and checked by reading, not compiled, not run on hardware
//
#include <hip/hip_runtime.h>
#include <hip/hip_bf16.h>

typedef __attribute__((ext_vector_type(16))) __bf16 v16bf;
typedef __attribute__((ext_vector_type(2)))  __bf16 v2bf;
typedef __attribute__((ext_vector_type(8)))  float  v8f;

#define TILE_M 256
#define TILE_N 128
#define TILE_K 32
#define K_DIM  4096
#define N_DIM  4096
#define LDS_STRIDE 40   // ushorts/row = 80B = 20 banks -> conflict-free b128 frag loads
#define ROW_STEP (32 * K_DIM * 4)   // byte stride between j-chunks: folds into IOFFSET

// exact RNE f32->bf16 bits (one-time setup: scale/bias)
__device__ __forceinline__ unsigned bfbits_rne(float f) {
    unsigned u = __builtin_bit_cast(unsigned, f);
    return (u + 0x7FFFu + ((u >> 16) & 1u)) >> 16;
}
__device__ __forceinline__ float bftrunc(float f) {
    return __builtin_bit_cast(float, bfbits_rne(f) << 16);
}
// fast packed f32x2 -> bf16x2: round-half-up + single v_perm_b32
__device__ __forceinline__ unsigned pk_bf16(float a, float b) {
#if __has_builtin(__builtin_amdgcn_cvt_pk_bf16_f32)
    v2bf r = __builtin_amdgcn_cvt_pk_bf16_f32(a, b);
    return __builtin_bit_cast(unsigned, r);
#else
    unsigned ua = __builtin_bit_cast(unsigned, a) + 0x8000u;
    unsigned ub = __builtin_bit_cast(unsigned, b) + 0x8000u;
    return __builtin_amdgcn_perm(ub, ua, 0x07060302u);  // {bf(b), bf(a)}
#endif
}

union FragU { v16bf v; uint4 q[2]; };

__global__ __launch_bounds__(256) void fp8linear_wmma_kernel(
    const float* __restrict__ x,      // [M, K] f32
    const float* __restrict__ w,      // [N, K] f32 (fp8-representable -> exact in bf16)
    const float* __restrict__ scale,  // [N]
    const float* __restrict__ bias,   // [N]
    __hip_bfloat16* __restrict__ out) // [M, N] bf16
{
    __shared__ __align__(16) unsigned short As[2][TILE_M][LDS_STRIDE]; // 40KB
    __shared__ __align__(16) unsigned short Bs[2][TILE_N][LDS_STRIDE]; // 20KB

    const int tid    = threadIdx.x;
    const int lane   = tid & 31;
    const int wv     = tid >> 5;
    const int waveM  = wv & 3;    // 4 waves along M -> 64 rows each
    const int waveN  = wv >> 2;   // 2 waves along N -> 64 cols each
    const int laneLo = lane & 15;
    const int laneHi = lane >> 4;

    const long gm0 = (long)blockIdx.y * TILE_M;
    const long gn0 = (long)blockIdx.x * TILE_N;

    // Single per-thread running pointer per matrix; j-chunks via constant IOFFSET.
    const int r0  = tid >> 3;        // 0..31 (j-chunk row base; row_j = r0 + 32*j)
    const int kc0 = (tid & 7) * 4;   // f32 column within the 32-wide K tile
    const char* aCur = (const char*)(x + (gm0 + r0) * (long)K_DIM + kc0);
    const char* bCur = (const char*)(w + (gn0 + r0) * (long)K_DIM + kc0);

    float sbJ[4];                    // bf16(scale[row]) hoisted out of the K loop
#pragma unroll
    for (int j = 0; j < 4; ++j) sbJ[j] = bftrunc(scale[gn0 + r0 + 32 * j]);

    float4 aReg[8], bReg[4];
    auto gload = [&]() {
#pragma unroll
        for (int j = 0; j < 8; ++j)
            aReg[j] = *(const float4*)(aCur + (long)j * ROW_STEP);  // imm offset
#pragma unroll
        for (int j = 0; j < 4; ++j)
            bReg[j] = *(const float4*)(bCur + (long)j * ROW_STEP);  // imm offset
        aCur += TILE_K * 4;   // one u64 add per matrix per iteration
        bCur += TILE_K * 4;
    };

    auto stage = [&](int p) {
#pragma unroll
        for (int j = 0; j < 8; ++j) {
            uint2 pa;
            pa.x = pk_bf16(aReg[j].x, aReg[j].y);
            pa.y = pk_bf16(aReg[j].z, aReg[j].w);
            *(uint2*)&As[p][r0 + 32 * j][kc0] = pa;   // 32*80B = imm ds offset
        }
#pragma unroll
        for (int j = 0; j < 4; ++j) {
            float sb = sbJ[j];
            uint2 pb;   // w exact in bf16; single rounding of (w * bf16(scale))
            pb.x = pk_bf16(bReg[j].x * sb, bReg[j].y * sb);
            pb.y = pk_bf16(bReg[j].z * sb, bReg[j].w * sb);
            *(uint2*)&Bs[p][r0 + 32 * j][kc0] = pb;
        }
    };

    v8f acc[4][4] = {};

    const int KT = K_DIM / TILE_K;
    gload();
    stage(0);
    __syncthreads();

    for (int kt = 0; kt < KT; ++kt) {
        const int p = kt & 1;
        if (kt + 1 < KT) gload();         // issue loads; wait deferred into stage()

        FragU aF[4], bF[4];
        // A frag (16-bit 16x32): lanes 0-15 -> K 0..7 & 16..23; lanes 16-31 -> K 8..15 & 24..31
#pragma unroll
        for (int mt = 0; mt < 4; ++mt) {
            int m    = waveM * 64 + mt * 16 + laneLo;
            int kOff = laneHi * 8;
            aF[mt].q[0] = *(const uint4*)&As[p][m][kOff];
            aF[mt].q[1] = *(const uint4*)&As[p][m][kOff + 16];
        }
        // B frag (32x16): lanes 0-15 -> K 0..7 & 16..23; lanes 16-31 -> K 8..15 & 24..31; N = lane%16
#pragma unroll
        for (int nt = 0; nt < 4; ++nt) {
            int n    = waveN * 64 + nt * 16 + laneLo;
            int kOff = laneHi * 8;
            bF[nt].q[0] = *(const uint4*)&Bs[p][n][kOff];
            bF[nt].q[1] = *(const uint4*)&Bs[p][n][kOff + 16];
        }
#pragma unroll
        for (int mt = 0; mt < 4; ++mt)
#pragma unroll
            for (int nt = 0; nt < 4; ++nt)
                acc[mt][nt] = __builtin_amdgcn_wmma_f32_16x16x32_bf16(
                    false, aF[mt].v, false, bF[nt].v,
                    (short)0, acc[mt][nt], false, false);

        if (kt + 1 < KT) stage(p ^ 1);    // overlaps load-wait with WMMA burst above
        __syncthreads();
    }

    // Epilogue: VGPR v -> M = v + 8*laneHi, N = laneLo; packed converts + NT stores
    unsigned short* outp = (unsigned short*)out;
#pragma unroll
    for (int nt = 0; nt < 4; ++nt) {
        long  n  = gn0 + waveN * 64 + nt * 16 + laneLo;
        float bb = bftrunc(bias[n]);
#pragma unroll
        for (int mt = 0; mt < 4; ++mt) {
            long m0 = gm0 + waveM * 64 + mt * 16 + laneHi * 8;
#pragma unroll
            for (int v = 0; v < 8; v += 2) {
                unsigned pkd = pk_bf16(acc[mt][nt][v] + bb, acc[mt][nt][v + 1] + bb);
                __builtin_nontemporal_store((unsigned short)(pkd & 0xFFFFu),
                                            &outp[(m0 + v) * N_DIM + n]);
                __builtin_nontemporal_store((unsigned short)(pkd >> 16),
                                            &outp[(m0 + v + 1) * N_DIM + n]);
            }
        }
    }
}

extern "C" void kernel_launch(void* const* d_in, const int* in_sizes, int n_in,
                              void* d_out, int out_size, void* d_ws, size_t ws_size,
                              hipStream_t stream) {
    const float* x  = (const float*)d_in[0];
    const float* w  = (const float*)d_in[1];
    const float* sc = (const float*)d_in[2];
    const float* bi = (const float*)d_in[3];
    __hip_bfloat16* out = (__hip_bfloat16*)d_out;

    const int M = in_sizes[0] / K_DIM;  // B*S = 4096
    dim3 grid(N_DIM / TILE_N, M / TILE_M);
    fp8linear_wmma_kernel<<<grid, 256, 0, stream>>>(x, w, sc, bi, out);
}
